// MultiHeadAttention_39367670235344
// MI455X (gfx1250) — compile-verified, matrix operand order checked by reading, not run on hardware
//
#include <hip/hip_runtime.h>
#include <hip/hip_bf16.h>
#include <math.h>

typedef __attribute__((ext_vector_type(16))) _Float16 v16h;
typedef __attribute__((ext_vector_type(8)))  _Float16 v8h;
typedef __attribute__((ext_vector_type(8)))  float    v8f;

#define LOG2E 1.4426950408889634f
#define LDSPAD 40   // 32 f16 cols padded to 40 (80B rows -> bank-conflict-free B128 frag reads)

union Frag16 { v16h v; v8h x[2]; };

// D = A(16x32) * B(32x16) + C, f16 inputs, f32 accum
__device__ __forceinline__ v8f wmma32(v16h a, v16h b, v8f c) {
    return __builtin_amdgcn_wmma_f32_16x16x32_f16(
        /*neg_a=*/false, a, /*neg_b=*/false, b,
        /*c_mod=*/(short)0, c, /*reuse_a=*/false, /*reuse_b=*/false);
}

// Load a 16(row) x 32(K) f16 fragment from row-major memory (global or LDS).
// Per-lane: row = lane%16, contiguous 16B at col 8*(lane/16); K=16..31 at +16.
__device__ __forceinline__ v16h load_frag32(const _Float16* __restrict__ p0, int ld) {
    const int lane = threadIdx.x & 31;
    const int r    = lane & 15;
    const int c    = (lane >> 4) << 3;
    const _Float16* p = p0 + (size_t)r * (size_t)ld + c;
    Frag16 f;
    f.x[0] = *(const v8h*)(p);
    f.x[1] = *(const v8h*)(p + 16);
    return f.v;
}

// Async-copy one 128-row x 32-col f16 tile (row stride ldg elements) into a
// padded LDS tile. 256 threads, 2 x 16B chunks each. Tracked by ASYNCcnt.
__device__ __forceinline__ void async_tile_128x32(const _Float16* __restrict__ g,
                                                  _Float16* lds, int ldg) {
    const int t = threadIdx.x;
#pragma unroll
    for (int i = 0; i < 2; ++i) {
        int idx = t + 256 * i;          // 0..511 chunks
        int rr = idx >> 2, cc = idx & 3;
        const void* ga = (const void*)(g + (size_t)rr * ldg + cc * 8);
        unsigned la = (unsigned)(size_t)(lds + rr * LDSPAD + cc * 8);
        asm volatile("global_load_async_to_lds_b128 %0, %1, off"
                     :: "v"(la), "v"(ga) : "memory");
    }
}

__device__ __forceinline__ void wait_async_all() {
    asm volatile("s_wait_asynccnt 0x0" ::: "memory");
}

__global__ void cvt_f16(const float* __restrict__ in, _Float16* __restrict__ out, int n) {
    int i = blockIdx.x * blockDim.x + threadIdx.x;
    int stride = gridDim.x * blockDim.x;
    for (; i < n; i += stride) out[i] = (_Float16)in[i];
}

// ---------------- QKV projection: y = x @ W^T, 3 weights via blockIdx.z ----
// X: [4096,1024] f16. W*: [1024,1024] f16 (N x K row-major).
// Q,K out: [32][2048][64] f16 ; Vt out: [32][64][2048] f16 (transposed).
__global__ __launch_bounds__(256) void gemm_qkv(
    const _Float16* __restrict__ X,
    const _Float16* __restrict__ Wq,
    const _Float16* __restrict__ Wk,
    const _Float16* __restrict__ Wv,
    _Float16* __restrict__ Qo,
    _Float16* __restrict__ Ko,
    _Float16* __restrict__ Vt)
{
    __shared__ _Float16 sA[2][128 * LDSPAD] __attribute__((aligned(16)));
    __shared__ _Float16 sB[2][128 * LDSPAD] __attribute__((aligned(16)));

    const int wave = threadIdx.x >> 5;
    const int lane = threadIdx.x & 31;
    const int wm = wave >> 2, wn = wave & 3;           // 2x4 wave grid
    const int mB = blockIdx.x * 128;                   // block row base
    const int nB = blockIdx.y * 128;                   // block col base
    const int which = blockIdx.z;
    const _Float16* W = (which == 0) ? Wq : (which == 1) ? Wk : Wv;

    v8f z = {};
    v8f c[4][2];
#pragma unroll
    for (int i = 0; i < 4; ++i)
#pragma unroll
        for (int j = 0; j < 2; ++j) c[i][j] = z;

    // prologue: fill buffer 0
    async_tile_128x32(X + (size_t)mB * 1024, sA[0], 1024);
    async_tile_128x32(W + (size_t)nB * 1024, sB[0], 1024);
    wait_async_all();
    __syncthreads();

    int p = 0;
    for (int kk = 0; kk < 1024; kk += 32) {
        if (kk + 32 < 1024) {   // prefetch next tile into other buffer
            async_tile_128x32(X + (size_t)mB * 1024 + kk + 32, sA[p ^ 1], 1024);
            async_tile_128x32(W + (size_t)nB * 1024 + kk + 32, sB[p ^ 1], 1024);
        }
        v16h a[4], b[2];
#pragma unroll
        for (int i = 0; i < 4; ++i)
            a[i] = load_frag32(&sA[p][(wm * 64 + i * 16) * LDSPAD], LDSPAD);
#pragma unroll
        for (int j = 0; j < 2; ++j)
            b[j] = load_frag32(&sB[p][(wn * 32 + j * 16) * LDSPAD], LDSPAD);
#pragma unroll
        for (int i = 0; i < 4; ++i)
#pragma unroll
            for (int j = 0; j < 2; ++j)
                c[i][j] = wmma32(a[i], b[j], c[i][j]);
        wait_async_all();
        __syncthreads();
        p ^= 1;
    }

    const int r = lane & 15, hh = lane >> 4;
    const int m0 = mB + wm * 64, n0 = nB + wn * 32;
#pragma unroll
    for (int i = 0; i < 4; ++i)
#pragma unroll
        for (int j = 0; j < 2; ++j)
#pragma unroll
            for (int v = 0; v < 8; ++v) {
                int M = m0 + i * 16 + v + 8 * hh;      // global row (b*2048+s)
                int N = n0 + j * 16 + r;               // global col (h*64+hd)
                _Float16 val = (_Float16)c[i][j][v];
                int bh = ((M >> 11) << 4) + (N >> 6);
                int s  = M & 2047;
                int hd = N & 63;
                if (which < 2) {
                    _Float16* O = (which == 0) ? Qo : Ko;
                    O[((size_t)bh * 2048 + s) * 64 + hd] = val;
                } else {
                    Vt[((size_t)bh * 64 + hd) * 2048 + s] = val;
                }
            }
}

// ---------------- Flash attention: 1 wave == one 16-row q tile -------------
__global__ __launch_bounds__(256) void attn(
    const _Float16* __restrict__ Q,   // [32][2048][64]
    const _Float16* __restrict__ K,   // [32][2048][64]
    const _Float16* __restrict__ Vt,  // [32][64][2048]
    _Float16* __restrict__ Ctx)       // [4096][1024]
{
    __shared__ _Float16 sP[8][16][32] __attribute__((aligned(16)));

    const int wave = threadIdx.x >> 5;
    const int lane = threadIdx.x & 31;
    const int r = lane & 15, hh = lane >> 4;
    const int bh = blockIdx.x;                       // 0..31
    const int q0 = blockIdx.y * 128 + wave * 16;
    const int qend = q0 + 15;
    const int b_ = bh >> 4, h = bh & 15;

    const _Float16* Qb = Q  + (size_t)bh * 2048 * 64;
    const _Float16* Kb = K  + (size_t)bh * 2048 * 64;
    const _Float16* Vb = Vt + (size_t)bh * 64 * 2048;

    v16h qa[2];
#pragma unroll
    for (int f = 0; f < 2; ++f)
        qa[f] = load_frag32(Qb + (size_t)q0 * 64 + f * 32, 64);

    v8f z = {};
    v8f ctx[4];
#pragma unroll
    for (int f = 0; f < 4; ++f) ctx[f] = z;
    float mrow[8], lrow[8];
#pragma unroll
    for (int v = 0; v < 8; ++v) { mrow[v] = -__builtin_inff(); lrow[v] = 0.0f; }

    const float scale = 0.125f;  // 1/sqrt(64)

    for (int k0 = 0; k0 <= qend; k0 += 32) {
        // ---- scores 16(q) x 32(k): 4 WMMAs over hd = 64 ----
        v8f s0 = z, s1 = z;
#pragma unroll
        for (int f = 0; f < 2; ++f) {
            v16h kb0 = load_frag32(Kb + (size_t)(k0)      * 64 + f * 32, 64);
            v16h kb1 = load_frag32(Kb + (size_t)(k0 + 16) * 64 + f * 32, 64);
            s0 = wmma32(qa[f], kb0, s0);
            s1 = wmma32(qa[f], kb1, s1);
        }
        // ---- causal mask + online softmax in C-frag layout ----
#pragma unroll
        for (int v = 0; v < 8; ++v) {
            int qrow = q0 + v + 8 * hh;
            float e0 = (k0 + r      <= qrow) ? s0[v] * scale : -__builtin_inff();
            float e1 = (k0 + 16 + r <= qrow) ? s1[v] * scale : -__builtin_inff();
            float rm = fmaxf(e0, e1);
            rm = fmaxf(rm, __shfl_xor(rm, 1, 32));
            rm = fmaxf(rm, __shfl_xor(rm, 2, 32));
            rm = fmaxf(rm, __shfl_xor(rm, 4, 32));
            rm = fmaxf(rm, __shfl_xor(rm, 8, 32));
            float mnew = fmaxf(mrow[v], rm);
            float corr = exp2f((mrow[v] - mnew) * LOG2E);
            float p0 = exp2f((e0 - mnew) * LOG2E);
            float p1 = exp2f((e1 - mnew) * LOG2E);
            float rs = p0 + p1;
            rs += __shfl_xor(rs, 1, 32);
            rs += __shfl_xor(rs, 2, 32);
            rs += __shfl_xor(rs, 4, 32);
            rs += __shfl_xor(rs, 8, 32);
            lrow[v] = lrow[v] * corr + rs;
            mrow[v] = mnew;
#pragma unroll
            for (int f = 0; f < 4; ++f) ctx[f][v] *= corr;
            sP[wave][v + 8 * hh][r]      = (_Float16)p0;
            sP[wave][v + 8 * hh][16 + r] = (_Float16)p1;
        }
        // C-frag -> A-frag relayout via per-wave LDS tile (same-wave DS ops
        // are in-order; explicit wait + compiler fence for safety)
        asm volatile("s_wait_dscnt 0x0" ::: "memory");
        Frag16 pa;
        pa.x[0] = *(const v8h*)&sP[wave][r][hh * 8];
        pa.x[1] = *(const v8h*)&sP[wave][r][16 + hh * 8];
        // ---- ctx += P @ V : 4 WMMAs over hd tiles ----
#pragma unroll
        for (int f = 0; f < 4; ++f) {
            v16h vb = load_frag32(Vb + (size_t)(f * 16) * 2048 + k0, 2048);
            ctx[f] = wmma32(pa.v, vb, ctx[f]);
        }
    }

    // ---- normalize + store f16 ctx in [b*2048+s][h*64+hd] layout ----
#pragma unroll
    for (int v = 0; v < 8; ++v) {
        float inv = 1.0f / lrow[v];
        int qrow = q0 + v + 8 * hh;
        size_t rowbase = ((size_t)b_ * 2048 + qrow) * 1024 + h * 64;
#pragma unroll
        for (int f = 0; f < 4; ++f)
            Ctx[rowbase + f * 16 + r] = (_Float16)(ctx[f][v] * inv);
    }
}

// ---------------- Output projection: y = ctx @ Wo^T + bo (f32 out) --------
__global__ __launch_bounds__(256) void gemm_out(
    const _Float16* __restrict__ A,   // [4096,1024]
    const _Float16* __restrict__ W,   // [1024,1024]
    const float* __restrict__ bias,   // [1024]
    float* __restrict__ Y)            // [4096,1024]
{
    __shared__ _Float16 sA[2][128 * LDSPAD] __attribute__((aligned(16)));
    __shared__ _Float16 sB[2][128 * LDSPAD] __attribute__((aligned(16)));

    const int wave = threadIdx.x >> 5;
    const int lane = threadIdx.x & 31;
    const int wm = wave >> 2, wn = wave & 3;
    const int mB = blockIdx.x * 128;
    const int nB = blockIdx.y * 128;

    v8f z = {};
    v8f c[4][2];
#pragma unroll
    for (int i = 0; i < 4; ++i)
#pragma unroll
        for (int j = 0; j < 2; ++j) c[i][j] = z;

    async_tile_128x32(A + (size_t)mB * 1024, sA[0], 1024);
    async_tile_128x32(W + (size_t)nB * 1024, sB[0], 1024);
    wait_async_all();
    __syncthreads();

    int p = 0;
    for (int kk = 0; kk < 1024; kk += 32) {
        if (kk + 32 < 1024) {
            async_tile_128x32(A + (size_t)mB * 1024 + kk + 32, sA[p ^ 1], 1024);
            async_tile_128x32(W + (size_t)nB * 1024 + kk + 32, sB[p ^ 1], 1024);
        }
        v16h a[4], b[2];
#pragma unroll
        for (int i = 0; i < 4; ++i)
            a[i] = load_frag32(&sA[p][(wm * 64 + i * 16) * LDSPAD], LDSPAD);
#pragma unroll
        for (int j = 0; j < 2; ++j)
            b[j] = load_frag32(&sB[p][(wn * 32 + j * 16) * LDSPAD], LDSPAD);
#pragma unroll
        for (int i = 0; i < 4; ++i)
#pragma unroll
            for (int j = 0; j < 2; ++j)
                c[i][j] = wmma32(a[i], b[j], c[i][j]);
        wait_async_all();
        __syncthreads();
        p ^= 1;
    }

    const int r = lane & 15, hh = lane >> 4;
    const int m0 = mB + wm * 64, n0 = nB + wn * 32;
#pragma unroll
    for (int i = 0; i < 4; ++i)
#pragma unroll
        for (int j = 0; j < 2; ++j)
#pragma unroll
            for (int v = 0; v < 8; ++v) {
                int M = m0 + i * 16 + v + 8 * hh;
                int N = n0 + j * 16 + r;
                Y[(size_t)M * 1024 + N] = c[i][j][v] + bias[N];
            }
}

extern "C" void kernel_launch(void* const* d_in, const int* in_sizes, int n_in,
                              void* d_out, int out_size, void* d_ws, size_t ws_size,
                              hipStream_t stream) {
    (void)in_sizes; (void)n_in; (void)out_size; (void)ws_size;
    const float* x  = (const float*)d_in[0];
    const float* Wq = (const float*)d_in[1];
    const float* Wk = (const float*)d_in[2];
    const float* Wv = (const float*)d_in[3];
    const float* Wo = (const float*)d_in[4];
    const float* bo = (const float*)d_in[5];

    char* ws = (char*)d_ws;
    const size_t MB = 1u << 20;
    _Float16* xh   = (_Float16*)(ws + 0);          // 8 MB (reused for ctx later)
    _Float16* Wqh  = (_Float16*)(ws + 8  * MB);    // 2 MB
    _Float16* Wkh  = (_Float16*)(ws + 10 * MB);
    _Float16* Wvh  = (_Float16*)(ws + 12 * MB);
    _Float16* Woh  = (_Float16*)(ws + 14 * MB);
    _Float16* Qh   = (_Float16*)(ws + 16 * MB);    // 8 MB
    _Float16* Kh   = (_Float16*)(ws + 24 * MB);    // 8 MB
    _Float16* Vth  = (_Float16*)(ws + 32 * MB);    // 8 MB
    _Float16* Ctxh = (_Float16*)(ws + 0);          // reuse xh region (dead after QKV GEMM)

    cvt_f16<<<512, 256, 0, stream>>>(x,  xh,  4096 * 1024);
    cvt_f16<<<256, 256, 0, stream>>>(Wq, Wqh, 1024 * 1024);
    cvt_f16<<<256, 256, 0, stream>>>(Wk, Wkh, 1024 * 1024);
    cvt_f16<<<256, 256, 0, stream>>>(Wv, Wvh, 1024 * 1024);
    cvt_f16<<<256, 256, 0, stream>>>(Wo, Woh, 1024 * 1024);

    gemm_qkv<<<dim3(32, 8, 3), 256, 0, stream>>>(xh, Wqh, Wkh, Wvh, Qh, Kh, Vth);
    attn<<<dim3(32, 16), 256, 0, stream>>>(Qh, Kh, Vth, Ctxh);
    gemm_out<<<dim3(32, 8), 256, 0, stream>>>(Ctxh, Woh, bo, (float*)d_out);
}
